// FasterRCNN_89240830476482
// MI455X (gfx1250) — matrix operand order repaired and checked by reading, not compiled, not run on hardware
//
#include <hip/hip_runtime.h>

#define IMGF   800.0f
#define NANCH  22500
#define RL     128
#define ROI_S  7
#define CCH    512
#define FLAT   25088      // 7*7*512
#define KCONV  4608       // 9*512
#define KT_N   144        // KCONV/32 k-tiles
#define FHW    50
#define NCLS   21

typedef __attribute__((ext_vector_type(16))) _Float16 v16h;
typedef __attribute__((ext_vector_type(8)))  _Float16 v8h;
typedef __attribute__((ext_vector_type(8)))  float    v8f;
typedef __attribute__((ext_vector_type(2)))  float    v2f;
typedef __attribute__((ext_vector_type(4)))  int      v4i;

#define GLOBAL_AS __attribute__((address_space(1)))
#define LDS_AS    __attribute__((address_space(3)))

#if __has_builtin(__builtin_amdgcn_global_load_async_to_lds_b128) && \
    __has_builtin(__builtin_amdgcn_s_wait_asynccnt)
#define USE_ASYNC_COPY 1
#endif

__device__ __forceinline__ float relu_f(float v)  { return v > 0.f ? v : 0.f; }
__device__ __forceinline__ float leaky_f(float v) { return v > 0.f ? v : 0.1f * v; }

// ---------------------------------------------------------------- decode
__global__ void decode_kernel(const float* __restrict__ anchors,
                              const float* __restrict__ deltas,
                              const float* __restrict__ scores,
                              float* __restrict__ boxes,
                              float* __restrict__ swork) {
  int i = blockIdx.x * blockDim.x + threadIdx.x;
  if (i >= NANCH) return;
  float a0 = anchors[i*4+0], a1 = anchors[i*4+1];
  float a2 = anchors[i*4+2], a3 = anchors[i*4+3];
  float ah = a2 - a0, aw = a3 - a1;
  float acy = (a0 + a2) * 0.5f, acx = (a1 + a3) * 0.5f;
  float d0 = deltas[i*4+0], d1 = deltas[i*4+1];
  float d2 = fminf(fmaxf(deltas[i*4+2], -4.f), 4.f);
  float d3 = fminf(fmaxf(deltas[i*4+3], -4.f), 4.f);
  float cy = d0 * ah + acy, cx = d1 * aw + acx;
  float h = __expf(d2) * ah, w = __expf(d3) * aw;
  boxes[i*4+0] = fminf(fmaxf(cy - h*0.5f, 0.f), IMGF);
  boxes[i*4+1] = fminf(fmaxf(cx - w*0.5f, 0.f), IMGF);
  boxes[i*4+2] = fminf(fmaxf(cy + h*0.5f, 0.f), IMGF);
  boxes[i*4+3] = fminf(fmaxf(cx + w*0.5f, 0.f), IMGF);
  float s = scores[i];
  swork[i] = (s > 0.0f) ? s : -__builtin_inff();
}

// ---------------------------------------------------------------- NMS (1 block)
__global__ void nms_kernel(const float* __restrict__ boxes,
                           float* __restrict__ s,
                           float* __restrict__ prop,
                           float* __restrict__ valid_out) {
  __shared__ float sv[256];
  __shared__ int   si[256];
  __shared__ float sbox[4];
  int tid = threadIdx.x;
  for (int it = 0; it < RL; ++it) {
    float bv = -__builtin_inff(); int bi = 0;
    for (int j = tid; j < NANCH; j += 256) {
      float v = s[j];
      if (v > bv) { bv = v; bi = j; }
    }
    sv[tid] = bv; si[tid] = bi;
    __syncthreads();
    for (int off = 128; off; off >>= 1) {
      if (tid < off) {
        if (sv[tid+off] > sv[tid] ||
            (sv[tid+off] == sv[tid] && si[tid+off] < si[tid])) {
          sv[tid] = sv[tid+off]; si[tid] = si[tid+off];
        }
      }
      __syncthreads();
    }
    if (tid == 0) {
      int idx = si[0];
      bool valid = sv[0] > -__builtin_inff();
      float sc = valid ? (1.0f / 16.0f) : 0.0f;
      #pragma unroll
      for (int k = 0; k < 4; ++k) {
        float bk = boxes[idx*4+k];
        sbox[k] = bk;
        prop[it*4+k] = bk * sc;
      }
      valid_out[it] = valid ? 1.0f : 0.0f;
    }
    __syncthreads();
    float by1 = sbox[0], bx1 = sbox[1], by2 = sbox[2], bx2 = sbox[3];
    float areaB = fmaxf(by2-by1, 0.f) * fmaxf(bx2-bx1, 0.f);
    for (int j = tid; j < NANCH; j += 256) {
      float oy1 = boxes[j*4+0], ox1 = boxes[j*4+1];
      float oy2 = boxes[j*4+2], ox2 = boxes[j*4+3];
      float yy1 = fmaxf(by1, oy1), xx1 = fmaxf(bx1, ox1);
      float yy2 = fminf(by2, oy2), xx2 = fminf(bx2, ox2);
      float inter = fmaxf(yy2-yy1, 0.f) * fmaxf(xx2-xx1, 0.f);
      float areaJ = fmaxf(oy2-oy1, 0.f) * fmaxf(ox2-ox1, 0.f);
      float iou = inter / fmaxf(areaB + areaJ - inter, 1e-8f);
      if (iou >= 0.7f) s[j] = -__builtin_inff();
    }
    __threadfence_block();
    __syncthreads();
  }
}

// ---------------------------------------------------------------- ROI align -> pre-activated f16
__global__ void roi_align_kernel(const float* __restrict__ feat,
                                 const float* __restrict__ prop,
                                 _Float16* __restrict__ rois_relu,
                                 _Float16* __restrict__ rois_leaky) {
  int r = blockIdx.x;
  float y1 = prop[r*4+0], x1 = prop[r*4+1];
  float y2 = prop[r*4+2], x2 = prop[r*4+3];
  for (int idx = threadIdx.x; idx < FLAT; idx += blockDim.x) {
    int p = idx >> 9, c = idx & 511;
    int i = p / ROI_S, j = p - i * ROI_S;
    float ys = y1 + (y2 - y1) * ((float)i * (1.0f/6.0f));
    float xs = x1 + (x2 - x1) * ((float)j * (1.0f/6.0f));
    float yf = fminf(fmaxf(floorf(ys), 0.f), 49.f);
    float xf = fminf(fmaxf(floorf(xs), 0.f), 49.f);
    int y0 = (int)yf, x0 = (int)xf;
    int y1i = min(y0 + 1, 49), x1i = min(x0 + 1, 49);
    float wy = fminf(fmaxf(ys - yf, 0.f), 1.f);
    float wx = fminf(fmaxf(xs - xf, 0.f), 1.f);
    float f00 = feat[(y0 *FHW + x0 )*CCH + c];
    float f01 = feat[(y0 *FHW + x1i)*CCH + c];
    float f10 = feat[(y1i*FHW + x0 )*CCH + c];
    float f11 = feat[(y1i*FHW + x1i)*CCH + c];
    float v = f00*(1.f-wy)*(1.f-wx) + f01*(1.f-wy)*wx
            + f10*wy*(1.f-wx)       + f11*wy*wx;
    rois_relu [(size_t)r*FLAT + idx] = (_Float16)relu_f(v);
    rois_leaky[(size_t)r*FLAT + idx] = (_Float16)leaky_f(v);
  }
}

// ---------------------------------------------------------------- pack conv weights into B-fragment layout (f16)
// wp[((ntile*144 + ktile)*32 + lane)*16 + j] = w[k][n]
__global__ void pack_w_kernel(const float* __restrict__ w, _Float16* __restrict__ wp) {
  int o = blockIdx.x * 256 + threadIdx.x;
  if (o >= KCONV * CCH) return;
  int j = o & 15, l = (o >> 4) & 31;
  int rest = o >> 9;
  int kt = rest % KT_N, ntl = rest / KT_N;
  int k = kt * 32 + ((j >= 8) ? 16 : 0) + ((l >= 16) ? 8 : 0) + (j & 7);
  int n = ntl * 16 + (l & 15);
  wp[o] = (_Float16)w[(size_t)k * CCH + n];
}

// ---------------------------------------------------------------- 3x3 conv, f16 WMMA implicit GEMM
// input: pre-activated f16 [roi][49][512]; weights: packed fragments.
// OUT_RELU_F16: 1 -> write relu(out) as f16 (feeds next conv); 0 -> raw f32.
template <int OUT_RELU_F16>
__global__ __launch_bounds__(128) void conv3x3_wmma_kernel(
    const _Float16* __restrict__ inh, const _Float16* __restrict__ wp,
    const float* __restrict__ bias, float* __restrict__ outf,
    _Float16* __restrict__ outh) {
  __shared__ _Float16 sA[FLAT];   // 50176 B
  int roi = blockIdx.y;
  int nbase = blockIdx.x * 64;
  const _Float16* src = inh + (size_t)roi * FLAT;

#ifdef USE_ASYNC_COPY
  for (int i = threadIdx.x; i < FLAT / 8; i += 128) {
    __builtin_amdgcn_global_load_async_to_lds_b128(
        (GLOBAL_AS v4i*)(_Float16*)(src + (size_t)i * 8),
        (LDS_AS v4i*)(&sA[i * 8]), 0, 0);
  }
  __builtin_amdgcn_s_wait_asynccnt(0);
  __syncthreads();
#else
  for (int i = threadIdx.x; i < FLAT / 8; i += 128)
    *(v8h*)&sA[i * 8] = *(const v8h*)&src[i * 8];
  __syncthreads();
#endif

  int wave = threadIdx.x >> 5, lane = threadIdx.x & 31;
  int m0 = wave * 16;
  int mm = m0 + (lane & 15);
  int yy = mm / ROI_S, xx = mm - yy * ROI_S;
  bool mok = mm < 49;
  int kext = (lane >> 4) * 8;         // +8 element offset for upper half-wave
  int nlane = lane & 15;
  const _Float16* wpl = wp + (size_t)(nbase >> 4) * KT_N * 512 + lane * 16;
  v8f acc[4] = {};
  v8h zero8 = {};

  #pragma unroll
  for (int tap = 0; tap < 9; ++tap) {
    int iy = yy + tap / 3 - 1;
    int ix = xx + (tap - (tap / 3) * 3) - 1;
    bool ok = mok && ((unsigned)iy < 7u) && ((unsigned)ix < 7u);
    int abase = (iy * ROI_S + ix) * CCH + kext;
    for (int cs = 0; cs < 16; ++cs) {
      union { v16h v; v8h h[2]; } au;
      if (ok) {
        au.h[0] = *(const v8h*)&sA[abase + cs * 32];
        au.h[1] = *(const v8h*)&sA[abase + cs * 32 + 16];
      } else {
        au.h[0] = zero8; au.h[1] = zero8;
      }
      int kt = tap * 16 + cs;
      const _Float16* bp = wpl + (size_t)kt * 512;
      #pragma unroll
      for (int nt = 0; nt < 4; ++nt) {
        v16h b = *(const v16h*)(bp + (size_t)nt * KT_N * 512);
        acc[nt] = __builtin_amdgcn_wmma_f32_16x16x32_f16(
            false, au.v, false, b, (short)0, acc[nt], false, false);
      }
    }
  }

  int mw = m0 + ((lane >= 16) ? 8 : 0);
  #pragma unroll
  for (int nt = 0; nt < 4; ++nt) {
    int n = nbase + nt * 16 + nlane;
    float bb = bias[n];
    #pragma unroll
    for (int rrow = 0; rrow < 8; ++rrow) {
      int M = mw + rrow;
      if (M < 49) {
        float val = acc[nt][rrow] + bb;
        size_t off = (size_t)roi * FLAT + (size_t)M * CCH + n;
        if (OUT_RELU_F16) outh[off] = (_Float16)relu_f(val);
        else              outf[off] = val;
      }
    }
  }
}

// ---------------------------------------------------------------- dense GEMM, native f32 WMMA (memory-bound layers)
template <int ACTA>
__global__ void dense_f32_wmma_kernel(const float* __restrict__ x,
                                      const float* __restrict__ wd,
                                      const float* __restrict__ bias,
                                      float* __restrict__ y,
                                      int Kdim, int Ndim, int ntilesN) {
  int t = blockIdx.x * (blockDim.x >> 5) + (threadIdx.x >> 5);
  int lane = threadIdx.x & 31;
  int mt = t / ntilesN, nt = t - mt * ntilesN;
  int m = mt * 16 + (lane & 15);
  int n = nt * 16 + (lane & 15);
  bool nok = n < Ndim;
  int kup = (lane >= 16) ? 2 : 0;
  v8f acc = {};
  int ksteps = Kdim >> 2;
  for (int ks = 0; ks < ksteps; ++ks) {
    int k0 = ks * 4 + kup;
    v2f a, b;
    #pragma unroll
    for (int j = 0; j < 2; ++j) {
      float av = x[(size_t)m * Kdim + k0 + j];
      if (ACTA) av = leaky_f(av);
      a[j] = av;
      b[j] = nok ? wd[(size_t)(k0 + j) * Ndim + n] : 0.f;
    }
    acc = __builtin_amdgcn_wmma_f32_16x16x4_f32(
        false, a, false, b, (short)0, acc, false, false);
  }
  if (nok) {
    float bb = bias[n];
    int mrow = mt * 16 + ((lane >= 16) ? 8 : 0);
    #pragma unroll
    for (int r = 0; r < 8; ++r)
      y[(size_t)(mrow + r) * Ndim + n] = acc[r] + bb;
  }
}

// ---------------------------------------------------------------- softmax + bbox head
__global__ void finalize_kernel(const float* __restrict__ logits,
                                const float* __restrict__ y1,
                                const float* __restrict__ wd2,
                                const float* __restrict__ b2,
                                float* __restrict__ out) {
  int r = threadIdx.x;          // 128 threads
  float mx = -__builtin_inff();
  for (int c = 0; c < NCLS; ++c) mx = fmaxf(mx, logits[r*NCLS + c]);
  float e[NCLS]; float sum = 0.f;
  for (int c = 0; c < NCLS; ++c) { e[c] = __expf(logits[r*NCLS + c] - mx); sum += e[c]; }
  float inv = 1.0f / sum;
  for (int c = 0; c < NCLS; ++c) out[r*25 + c] = e[c] * inv;
  for (int c = 0; c < 4; ++c) {
    float d = b2[c];
    for (int k = 0; k < 1024; ++k) d += y1[(size_t)r*1024 + k] * wd2[k*4 + c];
    out[r*25 + 21 + c] = d;
  }
}

// ---------------------------------------------------------------- launch
extern "C" void kernel_launch(void* const* d_in, const int* in_sizes, int n_in,
                              void* d_out, int out_size, void* d_ws, size_t ws_size,
                              hipStream_t stream) {
  const float* features  = (const float*)d_in[0];
  const float* rpn_scores= (const float*)d_in[1];
  const float* rpn_deltas= (const float*)d_in[2];
  const float* anchors   = (const float*)d_in[3];
  const float* cls_w1 = (const float*)d_in[4];  const float* cls_b1 = (const float*)d_in[5];
  const float* cls_w2 = (const float*)d_in[6];  const float* cls_b2 = (const float*)d_in[7];
  const float* cls_wd = (const float*)d_in[8];  const float* cls_bd = (const float*)d_in[9];
  const float* loc_w1 = (const float*)d_in[10]; const float* loc_b1 = (const float*)d_in[11];
  const float* loc_wd1= (const float*)d_in[12]; const float* loc_bd1= (const float*)d_in[13];
  const float* loc_wd2= (const float*)d_in[14]; const float* loc_bd2= (const float*)d_in[15];
  float* out = (float*)d_out;

  float* ws = (float*)d_ws;
  float* w_boxes  = ws; ws += (size_t)NANCH * 4;
  float* w_scores = ws; ws += NANCH;
  float* w_prop   = ws; ws += RL * 4;
  _Float16* w_roisA = (_Float16*)ws; ws += (size_t)RL * FLAT / 2;  // relu f16
  _Float16* w_roisB = (_Float16*)ws; ws += (size_t)RL * FLAT / 2;  // leaky f16
  _Float16* w_c1h   = (_Float16*)ws; ws += (size_t)RL * FLAT / 2;  // conv1 relu f16
  float* w_c2     = ws; ws += (size_t)RL * FLAT;                   // conv2 raw f32
  float* w_l1     = ws; ws += (size_t)RL * FLAT;                   // loc conv raw f32
  _Float16* wp1   = (_Float16*)ws; ws += (size_t)KCONV * CCH / 2;  // packed weights f16
  _Float16* wp2   = (_Float16*)ws; ws += (size_t)KCONV * CCH / 2;
  _Float16* wp3   = (_Float16*)ws; ws += (size_t)KCONV * CCH / 2;
  float* w_logits = ws; ws += RL * NCLS;
  float* w_y1     = ws; ws += (size_t)RL * 1024;

  int packGrid = (KCONV * CCH + 255) / 256;
  pack_w_kernel<<<packGrid, 256, 0, stream>>>(cls_w1, wp1);
  pack_w_kernel<<<packGrid, 256, 0, stream>>>(cls_w2, wp2);
  pack_w_kernel<<<packGrid, 256, 0, stream>>>(loc_w1, wp3);

  decode_kernel<<<(NANCH + 255) / 256, 256, 0, stream>>>(
      anchors, rpn_deltas, rpn_scores, w_boxes, w_scores);
  nms_kernel<<<1, 256, 0, stream>>>(w_boxes, w_scores, w_prop, out + RL * 25);
  roi_align_kernel<<<RL, 256, 0, stream>>>(features, w_prop, w_roisA, w_roisB);

  dim3 cgrid(8, RL);
  conv3x3_wmma_kernel<1><<<cgrid, 128, 0, stream>>>(w_roisA, wp1, cls_b1, nullptr, w_c1h);
  conv3x3_wmma_kernel<0><<<cgrid, 128, 0, stream>>>(w_c1h,   wp2, cls_b2, w_c2, nullptr);
  conv3x3_wmma_kernel<0><<<cgrid, 128, 0, stream>>>(w_roisB, wp3, loc_b1, w_l1, nullptr);

  dense_f32_wmma_kernel<0><<<2, 256, 0, stream>>>(w_c2, cls_wd, cls_bd, w_logits,
                                                  FLAT, NCLS, 2);
  dense_f32_wmma_kernel<1><<<64, 256, 0, stream>>>(w_l1, loc_wd1, loc_bd1, w_y1,
                                                   FLAT, 1024, 64);

  finalize_kernel<<<1, 128, 0, stream>>>(w_logits, w_y1, loc_wd2, loc_bd2, out);
}
